// L2VectorQuantizer_76209899700807
// MI455X (gfx1250) — matrix operand order repaired and checked by reading, not compiled, not run on hardware
//
#include <hip/hip_runtime.h>
#include <hip/hip_bf16.h>
#include <cstdint>
#include <cstddef>

// Problem constants (from reference)
#define NUM_CODE 8192
#define CODE_DIM 256
#define NB 16
#define HW 1024                     // 32*32
#define NROWS (NB * HW)             // 16384 z-vectors
#define ZELEMS ((size_t)NROWS * CODE_DIM)   // 4194304
#define LOSS_OFF (ZELEMS)
#define IDX_OFF  (ZELEMS + 1)

typedef __bf16 bf16_t;
typedef __attribute__((ext_vector_type(8)))  __bf16 v8bf;
typedef __attribute__((ext_vector_type(16))) __bf16 v16bf;
typedef __attribute__((ext_vector_type(8)))  float  v8f;

// ---- workspace layout (bytes); total ~25.2 MB ----
#define WS_EHI 0
#define WS_ELO (WS_EHI + (size_t)NUM_CODE * CODE_DIM * 2)
#define WS_ZHI (WS_ELO + (size_t)NUM_CODE * CODE_DIM * 2)
#define WS_ZLO (WS_ZHI + (size_t)NROWS * CODE_DIM * 2)
#define WS_EN  (WS_ZLO + (size_t)NROWS * CODE_DIM * 2)
#define WS_PART (WS_EN + (size_t)NUM_CODE * 4)

#define ROWP 264                 // LDS B-row pitch in bf16 (256 + 8 pad vs 512B-stride conflicts)
#define STAGE_B ((16 * ROWP) * 2)  // bytes per stage per matrix = 8448
#define NSTAGE 3
#define NTILE (NUM_CODE / 16)    // 512 code tiles of 16

__device__ __forceinline__ v8f wmma_bf16(v16bf a, v16bf b, v8f c) {
  return __builtin_amdgcn_wmma_f32_16x16x32_bf16(false, a, false, b, (short)0, c,
                                                 false, false);
}

__device__ __forceinline__ v16bf cat8(v8bf a, v8bf b) {
  return __builtin_shufflevector(a, b, 0, 1, 2, 3, 4, 5, 6, 7,
                                 8, 9, 10, 11, 12, 13, 14, 15);
}

// low 32 bits of a generic pointer to a __shared__ object == LDS byte offset
__device__ __forceinline__ uint32_t lds_u32(const void* p) {
  return (uint32_t)(uintptr_t)p;
}

// Issue one 16-code x 256-K bf16 tile (hi+lo) as 8 async global->LDS B128 copies.
// INST_OFFSET is added to BOTH the LDS and the global address (ISA 15.18.3), so
// imm offsets 0/16/32/48 cover this thread's 64B slice of each matrix.
__device__ __forceinline__ void issue_tile(uint32_t lh, uint32_t ll, uint32_t go,
                                           uint64_t gh, uint64_t gl) {
  asm volatile(
      "global_load_async_to_lds_b128 %0, %2, %3 offset:0\n\t"
      "global_load_async_to_lds_b128 %0, %2, %3 offset:16\n\t"
      "global_load_async_to_lds_b128 %0, %2, %3 offset:32\n\t"
      "global_load_async_to_lds_b128 %0, %2, %3 offset:48\n\t"
      "global_load_async_to_lds_b128 %1, %2, %4 offset:0\n\t"
      "global_load_async_to_lds_b128 %1, %2, %4 offset:16\n\t"
      "global_load_async_to_lds_b128 %1, %2, %4 offset:32\n\t"
      "global_load_async_to_lds_b128 %1, %2, %4 offset:48"
      :: "v"(lh), "v"(ll), "v"(go), "s"(gh), "s"(gl)
      : "memory");
}

// ---------- prep: embedding -> bf16 hi/lo split + ||e||^2 ----------
__global__ void prep_embed(const float* __restrict__ emb,
                           bf16_t* __restrict__ ehi, bf16_t* __restrict__ elo,
                           float* __restrict__ enorm) {
  const int code = blockIdx.x, t = threadIdx.x;
  const float x = emb[(size_t)code * CODE_DIM + t];
  const bf16_t h = (bf16_t)x;               // RNE
  const bf16_t l = (bf16_t)(x - (float)h);  // residual
  ehi[(size_t)code * CODE_DIM + t] = h;
  elo[(size_t)code * CODE_DIM + t] = l;

  float s = x * x;
  #pragma unroll
  for (int m = 16; m >= 1; m >>= 1) s += __shfl_xor(s, m, 32);
  __shared__ float red[8];
  if ((t & 31) == 0) red[t >> 5] = s;
  __syncthreads();
  if (t == 0) {
    float tot = 0.f;
    #pragma unroll
    for (int i = 0; i < 8; ++i) tot += red[i];
    enorm[code] = tot;
  }
}

// ---------- prep: z [b,c,hw] -> z_flat [n=(b,hw), c] bf16 hi/lo, LDS-tiled transpose ----------
__global__ void prep_z(const float* __restrict__ z,
                       bf16_t* __restrict__ zhi, bf16_t* __restrict__ zlo) {
  __shared__ float tile[32][33];
  const int b = blockIdx.z, hw0 = blockIdx.x * 32, c0 = blockIdx.y * 32;
  const int tx = threadIdx.x, ty = threadIdx.y;
  #pragma unroll
  for (int j = 0; j < 4; ++j) {
    const int cl = ty + j * 8;
    tile[cl][tx] = z[((size_t)(b * CODE_DIM + c0 + cl)) * HW + hw0 + tx];
  }
  __syncthreads();
  #pragma unroll
  for (int j = 0; j < 4; ++j) {
    const int hwl = ty + j * 8;
    const float x = tile[tx][hwl];
    const bf16_t h = (bf16_t)x;
    const bf16_t l = (bf16_t)(x - (float)h);
    const size_t o = ((size_t)(b * HW + hw0 + hwl)) * CODE_DIM + c0 + tx;
    zhi[o] = h;
    zlo[o] = l;
  }
}

// ---------- main: split-bf16 WMMA distance + argmin, async-LDS staged B ----------
// 256 blocks x 128 threads (4 waves). Each wave owns one 16-row WMMA M-tile and
// scans all 8192 codes in 16-wide tiles streamed into a 3-stage LDS ring via
// GLOBAL_LOAD_ASYNC_TO_LDS_B128 (ASYNCcnt), one barrier per tile.
__global__ __launch_bounds__(128) void vq_argmin(
    const bf16_t* __restrict__ zhi, const bf16_t* __restrict__ zlo,
    const bf16_t* __restrict__ ehi, const bf16_t* __restrict__ elo,
    const float* __restrict__ enorm,
    const float* __restrict__ emb, const float* __restrict__ zin,
    float* __restrict__ out, float* __restrict__ part) {
  __shared__ __align__(16) bf16_t sBh[NSTAGE][16 * ROWP];
  __shared__ __align__(16) bf16_t sBl[NSTAGE][16 * ROWP];
  __shared__ int sIdx[64];
  __shared__ float sRed[4];

  const int t = threadIdx.x;
  const int w = t >> 5, lane = t & 31;
  const int m = lane & 15, half = lane >> 4;
  const int rowbase = blockIdx.x * 64 + w * 16;

  // ---- preload A tiles (this wave's 16 rows, full K=256) per ISA 16-bit A layout:
  // element i -> K = k*32 + (i<8 ? half*8+i : 16+half*8+(i-8))
  v16bf AH[8], AL[8];
  {
    const bf16_t* za = zhi + (size_t)(rowbase + m) * CODE_DIM;
    const bf16_t* zb = zlo + (size_t)(rowbase + m) * CODE_DIM;
    #pragma unroll
    for (int k = 0; k < 8; ++k) {
      AH[k] = cat8(*(const v8bf*)(za + k * 32 + half * 8),
                   *(const v8bf*)(za + k * 32 + 16 + half * 8));
      AL[k] = cat8(*(const v8bf*)(zb + k * 32 + half * 8),
                   *(const v8bf*)(zb + k * 32 + 16 + half * 8));
    }
  }

  // async-copy mapping: thread covers 64 contiguous bytes of each 16x256 bf16 tile
  const int pcode = t >> 3;                         // code row 0..15
  const uint32_t slice = (uint32_t)((t & 7) * 64);  // byte slice within a 512B row
  const uint32_t gbase = (uint32_t)(pcode * 512) + slice;  // bytes into e-array (tile 0)
  const uint32_t lbh = lds_u32(&sBh[0][0]) + (uint32_t)(pcode * ROWP * 2) + slice;
  const uint32_t lbl = lds_u32(&sBl[0][0]) + (uint32_t)(pcode * ROWP * 2) + slice;
  const uint64_t gh = (uint64_t)(uintptr_t)ehi;
  const uint64_t gl = (uint64_t)(uintptr_t)elo;

  float minv[8];
  int mini[8];
  #pragma unroll
  for (int r = 0; r < 8; ++r) { minv[r] = 3.4e38f; mini[r] = 0; }

  const int n = m;  // this lane's code column within a tile (C/D layout: N = lane&15)

  // prologue: stages 0 and 1 in flight
  issue_tile(lbh, lbl, gbase, gh, gl);
  issue_tile(lbh + STAGE_B, lbl + STAGE_B, gbase + 8192u, gh, gl);

  int scur = 0;   // stage of current tile
  int snxt = 2;   // stage of tile+2

  for (int tile = 0; tile < NTILE; ++tile) {
    // own loads for this tile landed (in-order, 8 per tile, <=8 newer outstanding)
    if (tile < NTILE - 2) {
      asm volatile("s_wait_asynccnt 0x8" ::: "memory");
    } else {
      asm volatile("s_wait_asynccnt 0x0" ::: "memory");
    }
    __syncthreads();  // everyone's stage ready; everyone done reading stage (tile-1)%3

    if (tile + 2 < NTILE) {  // refill the stage read at tile-1 (safe after barrier)
      issue_tile(lbh + (uint32_t)snxt * STAGE_B, lbl + (uint32_t)snxt * STAGE_B,
                 gbase + (uint32_t)(tile + 2) * 8192u, gh, gl);
    }

    // two independent accumulator chains + double-buffered B regs: WMMAs alternate
    // chains while the next k's LDS loads are already in flight.
    const bf16_t* bBase = &sBh[scur][n * ROWP + half * 8];
    const bf16_t* lBase = &sBl[scur][n * ROWP + half * 8];
    v8f acc0 = {0.f, 0.f, 0.f, 0.f, 0.f, 0.f, 0.f, 0.f};
    v8f acc1 = {0.f, 0.f, 0.f, 0.f, 0.f, 0.f, 0.f, 0.f};
    v16bf BH = cat8(*(const v8bf*)(bBase), *(const v8bf*)(bBase + 16));
    v16bf BL = cat8(*(const v8bf*)(lBase), *(const v8bf*)(lBase + 16));
    #pragma unroll
    for (int k = 0; k < 8; ++k) {
      v16bf BHn, BLn;
      if (k < 7) {  // prefetch k+1's B while k's WMMAs execute
        BHn = cat8(*(const v8bf*)(bBase + (k + 1) * 32),
                   *(const v8bf*)(bBase + (k + 1) * 32 + 16));
        BLn = cat8(*(const v8bf*)(lBase + (k + 1) * 32),
                   *(const v8bf*)(lBase + (k + 1) * 32 + 16));
      }
      // split-bf16 fp32-emulation: z.e ~= zh.eh + zh.el + zl.eh
      acc0 = wmma_bf16(AH[k], BH, acc0);
      acc1 = wmma_bf16(AH[k], BL, acc1);
      acc0 = wmma_bf16(AL[k], BH, acc0);
      BH = BHn;
      BL = BLn;
    }
    const int code = tile * 16 + n;
    const float en = enorm[code];  // L0-resident, 64B per tile per wave
    #pragma unroll
    for (int r = 0; r < 8; ++r) {
      const float dot = acc0[r] + acc1[r];
      const float q = fmaf(-2.0f, dot, en);  // ||e||^2 - 2 z.e (||z||^2 row-constant)
      if (q < minv[r]) { minv[r] = q; mini[r] = code; }
    }

    scur = (scur == 2) ? 0 : scur + 1;
    snxt = (snxt == 2) ? 0 : snxt + 1;
  }

  // cross-lane argmin within each 16-lane group (same row M = half*8 + r, different N)
  #pragma unroll
  for (int s = 1; s < 16; s <<= 1) {
    #pragma unroll
    for (int r = 0; r < 8; ++r) {
      const float ov = __shfl_xor(minv[r], s, 32);
      const int oi = __shfl_xor(mini[r], s, 32);
      if (ov < minv[r] || (ov == minv[r] && oi < mini[r])) { minv[r] = ov; mini[r] = oi; }
    }
  }
  if (m == 0) {
    #pragma unroll
    for (int r = 0; r < 8; ++r) sIdx[w * 16 + half * 8 + r] = mini[r];
  }
  __syncthreads();

  // ---- output: indices, gathered z_q scattered back to NCHW, fused loss ----
  const int n0 = blockIdx.x * 64;
  const int b = n0 >> 10;
  const int hwb = n0 & 1023;
  const int rloc = t & 63;
  const int cpart = t >> 6;
  const int idx = sIdx[rloc];

  if (t < 64) out[IDX_OFF + n0 + t] = (float)sIdx[t];

  float lsum = 0.f;
  const float* erow = emb + (size_t)idx * CODE_DIM;
  #pragma unroll 4
  for (int cb = 0; cb < 128; ++cb) {
    const int c = cb * 2 + cpart;
    const float e = erow[c];
    const size_t off = ((size_t)(b * CODE_DIM + c)) * HW + hwb + rloc;  // NCHW, hw-contiguous
    const float zv = zin[off];
    out[off] = e;  // straight-through forward value == z_q
    const float d = e - zv;
    lsum = fmaf(d, d, lsum);
  }
  // loss = (1 + BETA) * mean(d^2); deterministic two-stage reduction (no atomics)
  #pragma unroll
  for (int s = 16; s >= 1; s >>= 1) lsum += __shfl_xor(lsum, s, 32);
  if (lane == 0) sRed[w] = lsum;
  __syncthreads();
  if (t == 0)
    part[blockIdx.x] = (sRed[0] + sRed[1] + sRed[2] + sRed[3]) * (1.25f / 4194304.0f);
}

__global__ void loss_reduce(const float* __restrict__ part, float* __restrict__ out) {
  float s = part[threadIdx.x];
  #pragma unroll
  for (int m = 16; m >= 1; m >>= 1) s += __shfl_xor(s, m, 32);
  __shared__ float red[8];
  if ((threadIdx.x & 31) == 0) red[threadIdx.x >> 5] = s;
  __syncthreads();
  if (threadIdx.x == 0) {
    float tot = 0.f;
    #pragma unroll
    for (int i = 0; i < 8; ++i) tot += red[i];
    out[LOSS_OFF] = tot;
  }
}

extern "C" void kernel_launch(void* const* d_in, const int* in_sizes, int n_in,
                              void* d_out, int out_size, void* d_ws, size_t ws_size,
                              hipStream_t stream) {
  const float* z = (const float*)d_in[0];    // [16, 256, 32, 32]
  const float* emb = (const float*)d_in[1];  // [8192, 256]
  float* out = (float*)d_out;                // z_st (4194304) | loss (1) | indices (16384)

  char* ws = (char*)d_ws;                    // needs ~25.2 MB
  bf16_t* ehi = (bf16_t*)(ws + WS_EHI);
  bf16_t* elo = (bf16_t*)(ws + WS_ELO);
  bf16_t* zhi = (bf16_t*)(ws + WS_ZHI);
  bf16_t* zlo = (bf16_t*)(ws + WS_ZLO);
  float* enorm = (float*)(ws + WS_EN);
  float* part = (float*)(ws + WS_PART);

  prep_embed<<<NUM_CODE, CODE_DIM, 0, stream>>>(emb, ehi, elo, enorm);
  prep_z<<<dim3(HW / 32, CODE_DIM / 32, NB), dim3(32, 8), 0, stream>>>(z, zhi, zlo);
  vq_argmin<<<NROWS / 64, 128, 0, stream>>>(zhi, zlo, ehi, elo, enorm, emb, z, out, part);
  loss_reduce<<<1, 256, 0, stream>>>(part, out);
}
